// VisualizeInternalGates_17892833755465
// MI455X (gfx1250) — matrix operand order repaired and checked by reading, not compiled, not run on hardware
//
#include <hip/hip_runtime.h>
#include <stdint.h>

#define VOCAB  82
#define EMBED  256
#define HIDDEN 128
#define BATCH  256
#define SEQ    512
#define NJ     384   // 3*HIDDEN (f | c | o)
#define WROW   (HIDDEN + EMBED)   // 384 columns per weight row

typedef __attribute__((ext_vector_type(16))) __bf16 v16bf;
typedef __attribute__((ext_vector_type(8)))  float  v8f;

__device__ __forceinline__ unsigned short f32_to_bf16_bits(float f) {
    union { float f; unsigned int u; } x; x.f = f;
    unsigned int u = x.u;
    u += 0x7FFFu + ((u >> 16) & 1u);   // round-to-nearest-even
    return (unsigned short)(u >> 16);
}

__device__ __forceinline__ float fast_sigmoid(float x) {
    return 1.0f / (1.0f + __expf(-x));
}
__device__ __forceinline__ float fast_tanh(float x) {
    return 1.0f - 2.0f / (__expf(2.0f * x) + 1.0f);
}

// ---------------------------------------------------------------------------
// Precompute 1: P[v][j] = emb[v,:] . W_g[r, H:H+E]   (82 x 384, f32)
// j = gate*128 + r ; gate 0=f, 1=c, 2=o.  Tiny (8 MFLOP), plain FMA.
// Collapses the whole T*B input projection (26 GFLOP, 200 MB) to a lookup.
// ---------------------------------------------------------------------------
__global__ void vig_precompute_P(const float* __restrict__ emb,
                                 const float* __restrict__ Wf,
                                 const float* __restrict__ Wc,
                                 const float* __restrict__ Wo,
                                 float* __restrict__ P) {
    int idx = blockIdx.x * blockDim.x + threadIdx.x;
    if (idx >= VOCAB * NJ) return;
    int v = idx / NJ;
    int j = idx % NJ;
    int gate = j / HIDDEN, r = j % HIDDEN;
    const float* W = (gate == 0) ? Wf : (gate == 1) ? Wc : Wo;
    const float* wrow = W + (size_t)r * WROW + HIDDEN;  // x-part of the row
    const float* erow = emb + (size_t)v * EMBED;
    float acc = 0.0f;
#pragma unroll 8
    for (int e = 0; e < EMBED; ++e) acc += erow[e] * wrow[e];
    P[idx] = acc;
}

// ---------------------------------------------------------------------------
// Precompute 2: recurrent weights, transposed + bf16 pair-packed.
// Whpk[kp][j] = pack( bf16(W_g[r][2kp]), bf16(W_g[r][2kp+1]) ), kp = 0..63
// B-matrix source for v_wmma_f32_16x16x32_bf16.
// ---------------------------------------------------------------------------
__global__ void vig_precompute_Wh(const float* __restrict__ Wf,
                                  const float* __restrict__ Wc,
                                  const float* __restrict__ Wo,
                                  unsigned int* __restrict__ Whpk) {
    int idx = blockIdx.x * blockDim.x + threadIdx.x;
    if (idx >= (HIDDEN / 2) * NJ) return;
    int kp = idx / NJ;
    int j  = idx % NJ;
    int gate = j / HIDDEN, r = j % HIDDEN;
    const float* W = (gate == 0) ? Wf : (gate == 1) ? Wc : Wo;
    float a = W[(size_t)r * WROW + 2 * kp];
    float b = W[(size_t)r * WROW + 2 * kp + 1];
    Whpk[(size_t)kp * NJ + j] =
        (unsigned int)f32_to_bf16_bits(a) | ((unsigned int)f32_to_bf16_bits(b) << 16);
}

// ---------------------------------------------------------------------------
// Recurrent kernel. One workgroup (8 waves, 256 thr) owns a 16-row batch
// tile for all 512 steps (rows are independent -> no cross-block sync).
// Wave w owns hidden cols [w*16, w*16+16) for ALL THREE gates (N-tiles
// w, w+8, w+16), so the gate nonlinearity + c update are pure register math.
// Recurrent weights live in 96 VGPRs per wave for the whole loop; h
// ping-pongs through two 4 KB LDS buffers -> ONE barrier per step.
// ---------------------------------------------------------------------------
__global__ void __launch_bounds__(256)
vig_lstm_kernel(const int* __restrict__ tokens,        // [B][T]
                const float* __restrict__ P,           // [82][384]
                const unsigned int* __restrict__ Whpk, // [64][384] bf16 pairs
                const float* __restrict__ clfW,        // [82][128]
                const float* __restrict__ clfb,        // [82]
                float* __restrict__ logits,            // [B][82]
                float* __restrict__ f_out,             // [T][B][H]
                float* __restrict__ o_out,
                float* __restrict__ ct_out) {
    __shared__ int          tok_lds[16][SEQ];             // 32 KB
    __shared__ unsigned int h_pk[2][16][HIDDEN / 2];      // 2 x 4 KB bf16 pairs
    __shared__ float        h_f32[16][HIDDEN];            // 8 KB (final h)

    const int tid  = threadIdx.x;
    const int lane = tid & 31;
    const int wv   = tid >> 5;           // wave id 0..7
    const int b0   = blockIdx.x * 16;    // batch-tile base

    // preload this tile's tokens (read-once: non-temporal) + zero h buffer 0
    for (int i = tid; i < 16 * SEQ; i += 256) {
        int r = i >> 9, t = i & (SEQ - 1);
        tok_lds[r][t] =
            __builtin_nontemporal_load(&tokens[(size_t)(b0 + r) * SEQ + t]);
    }
    for (int i = tid; i < 16 * (HIDDEN / 2); i += 256)
        ((unsigned int*)&h_pk[0][0][0])[i] = 0u;
    __syncthreads();

    const int nlo  = lane & 15;          // tile column (hidden) / A row (batch)
    const int half = lane >> 4;
    const int col  = wv * 16 + nlo;      // this lane's hidden index

    // ---- load B fragments once: 3 gates x 4 K-chunks x 8 VGPRs (resident) ----
    v16bf Bfr[3][4];
#pragma unroll
    for (int g = 0; g < 3; ++g) {
        const int jcol = g * HIDDEN + col;
#pragma unroll
        for (int kc = 0; kc < 4; ++kc) {
            int* bp = (int*)&Bfr[g][kc];
#pragma unroll
            for (int v = 0; v < 8; ++v) {
                // lanes 0-15: K=0..15 of chunk; lanes 16-31: K=16..31
                int kp = kc * 16 + half * 8 + v;
                bp[v] = (int)Whpk[(size_t)kp * NJ + jcol];
            }
        }
    }

    float creg[8], hnew[8];
#pragma unroll
    for (int v = 0; v < 8; ++v) { creg[v] = 0.0f; hnew[v] = 0.0f; }

    for (int t = 0; t < SEQ; ++t) {
        const int rb = t & 1;        // read h from this buffer
        const int wb = rb ^ 1;       // write next h into the other one

        // ---- A fragments (h) from LDS: row=lane&15, K split per lane half ----
        v16bf Afr[4];
#pragma unroll
        for (int kc = 0; kc < 4; ++kc) {
            const int kpbase = (kc * 32 + half * 16) >> 1;
            int* ap = (int*)&Afr[kc];
#pragma unroll
            for (int v = 0; v < 8; ++v) {
                int kp = kpbase + v;   // K +0..15
                ap[v] = (int)h_pk[rb][nlo][kp];
            }
        }

        // ---- accumulators seeded with x-projection via token table ----
        // lanes 0-15 of a wave share a token -> 16 consecutive cols, coalesced
        v8f acc[3];
#pragma unroll
        for (int g = 0; g < 3; ++g) {
            float* cp = (float*)&acc[g];
#pragma unroll
            for (int v = 0; v < 8; ++v) {
                int m   = v + 8 * half;                 // batch row in tile
                int tok = tok_lds[m][t];
                cp[v] = P[(size_t)tok * NJ + g * HIDDEN + col];
            }
        }

        // ---- S_g += h @ Wh_g^T : 12 WMMAs per step ----
#pragma unroll
        for (int kc = 0; kc < 4; ++kc) {
            acc[0] = __builtin_amdgcn_wmma_f32_16x16x32_bf16(
                false, Afr[kc], false, Bfr[0][kc], (short)0, acc[0], false, false);
            acc[1] = __builtin_amdgcn_wmma_f32_16x16x32_bf16(
                false, Afr[kc], false, Bfr[1][kc], (short)0, acc[1], false, false);
            acc[2] = __builtin_amdgcn_wmma_f32_16x16x32_bf16(
                false, Afr[kc], false, Bfr[2][kc], (short)0, acc[2], false, false);
        }

        // ---- gates + state update, all in registers; streaming NT stores ----
        const float* sf = (const float*)&acc[0];
        const float* sc = (const float*)&acc[1];
        const float* so = (const float*)&acc[2];
        const size_t outbase = ((size_t)t * BATCH + b0) * HIDDEN + col;
#pragma unroll
        for (int v = 0; v < 8; ++v) {
            int   m  = v + 8 * half;
            float f  = fast_sigmoid(sf[v]);
            float ct = fast_tanh(sc[v]);
            float o  = fast_sigmoid(so[v]);
            float cn = f * creg[v] + (1.0f - f) * ct;
            creg[v]  = cn;
            hnew[v]  = o * fast_tanh(cn);
            size_t off = outbase + (size_t)m * HIDDEN;
            __builtin_nontemporal_store(f,  &f_out[off]);
            __builtin_nontemporal_store(o,  &o_out[off]);
            __builtin_nontemporal_store(ct, &ct_out[off]);
        }

        // write next h into the other buffer (no conflict with this step's
        // reads); single barrier publishes it and protects buffer reuse.
        {
            unsigned short* h_us = (unsigned short*)&h_pk[wb][0][0];
#pragma unroll
            for (int v = 0; v < 8; ++v) {
                int m = v + 8 * half;
                h_us[m * HIDDEN + col] = f32_to_bf16_bits(hnew[v]);
            }
        }
        __syncthreads();
    }

    // ---- classifier tail: logits = h_T @ clf_W^T + clf_b (tiny) ----
#pragma unroll
    for (int v = 0; v < 8; ++v) {
        int m = v + 8 * half;
        h_f32[m][col] = hnew[v];
    }
    __syncthreads();
    for (int i = tid; i < 16 * VOCAB; i += 256) {
        int r = i / VOCAB;
        int vv = i % VOCAB;
        const float* wrow = clfW + (size_t)vv * HIDDEN;
        float acc = clfb[vv];
#pragma unroll 8
        for (int k = 0; k < HIDDEN; ++k) acc += h_f32[r][k] * wrow[k];
        logits[(size_t)(b0 + r) * VOCAB + vv] = acc;
    }
}

// ---------------------------------------------------------------------------
extern "C" void kernel_launch(void* const* d_in, const int* in_sizes, int n_in,
                              void* d_out, int out_size, void* d_ws, size_t ws_size,
                              hipStream_t stream) {
    const int*   tokens = (const int*)d_in[0];
    const float* emb    = (const float*)d_in[1];
    const float* Wf     = (const float*)d_in[2];
    const float* Wc     = (const float*)d_in[3];
    const float* Wo     = (const float*)d_in[4];
    const float* clfW   = (const float*)d_in[5];
    const float* clfb   = (const float*)d_in[6];

    float*        P    = (float*)d_ws;                               // 126 KB
    unsigned int* Whpk = (unsigned int*)((char*)d_ws + (256 << 10)); // 96 KB @ +256KB

    float* out    = (float*)d_out;
    float* logits = out;
    float* f_out  = out + (size_t)BATCH * VOCAB;
    float* o_out  = f_out + (size_t)SEQ * BATCH * HIDDEN;
    float* ct_out = o_out + (size_t)SEQ * BATCH * HIDDEN;

    vig_precompute_P<<<(VOCAB * NJ + 255) / 256, 256, 0, stream>>>(emb, Wf, Wc, Wo, P);
    vig_precompute_Wh<<<((HIDDEN / 2) * NJ + 255) / 256, 256, 0, stream>>>(Wf, Wc, Wo, Whpk);
    vig_lstm_kernel<<<BATCH / 16, 256, 0, stream>>>(tokens, P, Whpk, clfW, clfb,
                                                    logits, f_out, o_out, ct_out);
}
